// TCN_GCN_unit_6356551598679
// MI455X (gfx1250) — matrix operand order repaired and checked by reading, not compiled, not run on hardware
//
#include <hip/hip_runtime.h>
#include <math.h>

typedef _Float16 f16;
typedef __attribute__((ext_vector_type(16))) _Float16 v16h;
typedef __attribute__((ext_vector_type(8)))  _Float16 v8h;
typedef __attribute__((ext_vector_type(8)))  float    v8f;

#define N_    32
#define CIN_  64
#define COUT_ 64
#define T_    256
#define V_    25
#define VP_   32            // padded joint dim
#define G_    8
#define K_    3
#define D_    (COUT_ * K_)  // 192
#define P_    (T_ * V_)     // 6400 real positions
#define PP_   (T_ * VP_)    // 8192 padded positions
#define RSBN  0.99999500003749978f   // 1/sqrt(1 + 1e-5)

// per-lane K index for 16-bit B fragments (32x16): lanes 0-15 K={0..7,16..23}, 16-31 K={8..15,24..31}
static __device__ __forceinline__ int b_kidx(int hi, int j) { return j + (hi ? 8 : 0) + (j >= 8 ? 8 : 0); }

static __device__ __forceinline__ v16h concat8(v8h a, v8h b) {
  return __builtin_shufflevector(a, b, 0,1,2,3,4,5,6,7,8,9,10,11,12,13,14,15);
}
// A fragment (16-bit A 16x32): lane-lo K={0..7,16..23}, lane-hi K={8..15,24..31}
// -> two contiguous 16B runs at +0 and +32 bytes from (row + kk*32 + 8*hi)
static __device__ __forceinline__ v16h frag_a(const f16* row, int kk, int hi) {
  const f16* p = row + kk * 32 + (hi ? 8 : 0);
  return concat8(*(const v8h*)p, *(const v8h*)(p + 16));
}
// linear 16-half fragment (pre-swizzled buffers): 32B contiguous per lane
static __device__ __forceinline__ v16h frag_lin(const f16* p) {
  return concat8(*(const v8h*)p, *(const v8h*)(p + 8));
}

static __device__ __forceinline__ v8f wmma_f16(const v16h& a, const v16h& b, const v8f& c) {
  return __builtin_amdgcn_wmma_f32_16x16x32_f16(false, a, false, b, (short)0, c, false, false);
}

// ---------------------------------------------------------------------------
// normA[kg,v,w] = DecoupleA[kg,v,w] / (sum_v DecoupleA[kg,v,w] + 1e-3)
// ---------------------------------------------------------------------------
__global__ void k_norm_a(const float* __restrict__ dA, float* __restrict__ nA) {
  int idx = blockIdx.x * blockDim.x + threadIdx.x;
  if (idx >= K_ * G_ * V_) return;
  int w = idx % V_, kg = idx / V_;
  const float* base = dA + (size_t)kg * V_ * V_;
  float s = 0.f;
  for (int v = 0; v < V_; ++v) s += base[v * V_ + w];
  float inv = 1.f / (s + 0.001f);
  float* ob = nA + (size_t)kg * V_ * V_;
  for (int v = 0; v < V_; ++v) ob[v * V_ + w] = base[v * V_ + w] * inv;
}

// B-fragment prep: W [64 x 192] -> wbh[((kk*12+dtile)*32+lane)*16+j]
__global__ void k_prep_gemm_w(const float* __restrict__ W, f16* __restrict__ wbh) {
  int idx = blockIdx.x * blockDim.x + threadIdx.x;   // 2*12*32*16 = 12288
  if (idx >= 2 * 12 * 32 * 16) return;
  int j = idx & 15, lane = (idx >> 4) & 31, t3 = idx >> 9;
  int dtile = t3 % 12, kk = t3 / 12;
  int hi = lane >> 4, ln = lane & 15;
  int d = dtile * 16 + ln;
  int i = kk * 32 + b_kidx(hi, j);
  wbh[idx] = (f16)W[(size_t)i * D_ + d];
}

// B-fragment prep: normA -> nAh[(((kg)*2+wh)*32+lane)*16+j], zero-padded v,w >= 25
__global__ void k_prep_mp_b(const float* __restrict__ nA, f16* __restrict__ nAh) {
  int idx = blockIdx.x * blockDim.x + threadIdx.x;   // 24*2*32*16 = 24576
  if (idx >= K_ * G_ * 2 * 32 * 16) return;
  int j = idx & 15, lane = (idx >> 4) & 31, t3 = idx >> 9;
  int wh = t3 & 1, kg = t3 >> 1;
  int hi = lane >> 4, ln = lane & 15;
  int w = wh * 16 + ln;
  int v = b_kidx(hi, j);
  float val = (v < V_ && w < V_) ? nA[((size_t)kg * V_ + v) * V_ + w] : 0.f;
  nAh[idx] = (f16)val;
}

// B-fragment prep: conv_w [64][64][9] -> cwh[(((dt*2+kk)*4+nt)*32+lane)*16+j]
__global__ void k_prep_tcn_w(const float* __restrict__ cw, f16* __restrict__ cwh) {
  int idx = blockIdx.x * blockDim.x + threadIdx.x;   // 9*2*4*32*16 = 36864
  if (idx >= 9 * 2 * 4 * 32 * 16) return;
  int j = idx & 15, lane = (idx >> 4) & 31, t3 = idx >> 9;
  int nt = t3 & 3, t4 = t3 >> 2;
  int kk = t4 & 1, dt = t4 >> 1;
  int hi = lane >> 4, ln = lane & 15;
  int o = nt * 16 + ln;
  int i = kk * 32 + b_kidx(hi, j);
  cwh[idx] = (f16)cw[((size_t)o * 64 + i) * 9 + dt];
}

// zero the v=24..31 pad lanes of every yk row (v=24 is overwritten by the GEMM)
__global__ void k_zero_pad(f16* __restrict__ ykh) {
  size_t r = (size_t)blockIdx.x * blockDim.x + threadIdx.x;  // 32*3*64*256 rows
  v8h z = {};
  *(v8h*)(ykh + r * VP_ + 24) = z;
}

// ---------------------------------------------------------------------------
// GEMM: yk[n,k,c,t,v32] = BN0( x . W + b )  (f16, v padded to 32)
// block = (n, 64-position tile); 8 waves = 4 M-tiles x 2 N-halves (6 N-tiles)
// ---------------------------------------------------------------------------
__global__ __launch_bounds__(256)
void k_gcn_gemm(const float* __restrict__ x, const f16* __restrict__ wbh,
                const float* __restrict__ bconv, const float* __restrict__ g0,
                const float* __restrict__ b0, f16* __restrict__ ykh) {
  __shared__ f16 xt[64][72];    // [pos][in-ch], 144B pitch

  int n   = blockIdx.y;
  int pb  = blockIdx.x * 64;
  int tid = threadIdx.x;

  { // stage x: coalesced f32 reads along pos, packed b32 LDS stores
    int p  = tid & 63;
    int i2 = (tid >> 6) * 2;    // 0,2,4,6
    const float* xb = x + (size_t)n * CIN_ * P_ + pb + p;
    #pragma unroll
    for (int c = 0; c < 8; ++c) {
      int i = c * 8 + i2;
      union { f16 h[2]; unsigned u; } pk;
      pk.h[0] = (f16)xb[(size_t)i * P_];
      pk.h[1] = (f16)xb[(size_t)(i + 1) * P_];
      *(unsigned*)&xt[p][i] = pk.u;
    }
  }
  __syncthreads();

  int wave = tid >> 5, lane = tid & 31;
  int hi = lane >> 4, ln = lane & 15;
  int mb     = (wave & 3) * 16;
  int wave_n = wave >> 2;
  int arow   = mb + ln;

  v8f acc[6];
  #pragma unroll
  for (int t = 0; t < 6; ++t) acc[t] = (v8f){};

  #pragma unroll
  for (int kk = 0; kk < 2; ++kk) {
    v16h a = frag_a(&xt[arow][0], kk, hi);
    #pragma unroll
    for (int t = 0; t < 6; ++t) {
      int dtile = wave_n * 6 + t;
      v16h bf = frag_lin(wbh + ((size_t)(kk * 12 + dtile) * 32 + lane) * 16);
      acc[t] = wmma_f16(a, bf, acc[t]);
    }
  }

  #pragma unroll
  for (int t = 0; t < 6; ++t) {
    int d = wave_n * 96 + t * 16 + ln;
    float s0 = g0[d] * RSBN;
    float sh = bconv[d] * s0 + b0[d];
    int k = d >> 6, cc = d & 63;
    f16* ob = ykh + (((size_t)n * K_ + k) * COUT_ + cc) * PP_;
    #pragma unroll
    for (int r = 0; r < 8; ++r) {
      int pos = pb + mb + r + (hi ? 8 : 0);
      int tt = pos / 25, vv = pos - tt * 25;
      ob[(size_t)tt * VP_ + vv] = (f16)(acc[t][r] * s0 + sh);
    }
  }
}

// ---------------------------------------------------------------------------
// message passing: y1[n, t*32+w, c] = relu( BN1( sum_k yk[n,k,c,t,:] . nA[k,:,w] ) + x )
// one wave per (n, c, 16-row t-tile); all fragment loads are b128
// ---------------------------------------------------------------------------
__global__ __launch_bounds__(256)
void k_gcn_mp(const f16* __restrict__ ykh, const f16* __restrict__ nAh,
              const float* __restrict__ x, const float* __restrict__ g1,
              const float* __restrict__ b1, f16* __restrict__ y1t) {
  int n = blockIdx.z, c = blockIdx.y;
  int wave = threadIdx.x >> 5, lane = threadIdx.x & 31;
  int tb = (blockIdx.x * 8 + wave) * 16;
  int hi = lane >> 4, ln = lane & 15;
  int g  = c & 7;

  v8f acc[2];
  acc[0] = (v8f){}; acc[1] = (v8f){};
  int t = tb + ln;
  #pragma unroll
  for (int k = 0; k < 3; ++k) {
    const f16* ab = ykh + ((((size_t)n * K_ + k) * COUT_ + c) * T_ + t) * VP_;
    v16h a = frag_a(ab, 0, hi);                 // K = 32 padded v's
    int kg = k * G_ + g;
    v16h b0f = frag_lin(nAh + ((size_t)(kg * 2 + 0) * 32 + lane) * 16);
    v16h b1f = frag_lin(nAh + ((size_t)(kg * 2 + 1) * 32 + lane) * 16);
    acc[0] = wmma_f16(a, b0f, acc[0]);
    acc[1] = wmma_f16(a, b1f, acc[1]);
  }

  float s1 = g1[c] * RSBN;
  float sh = b1[c];
  const float* xb = x + ((size_t)n * COUT_ + c) * P_;
  f16* ob = y1t + (size_t)n * PP_ * COUT_ + c;
  #pragma unroll
  for (int wh = 0; wh < 2; ++wh) {
    int w = wh * 16 + ln;
    if (w < V_) {
      #pragma unroll
      for (int r = 0; r < 8; ++r) {
        int tt = tb + r + (hi ? 8 : 0);
        float val = acc[wh][r] * s1 + sh + xb[tt * V_ + w];
        ob[(size_t)(tt * VP_ + w) * COUT_] = (f16)(val > 0.f ? val : 0.f);
      }
    }
  }
}

// ---------------------------------------------------------------------------
// TCN: out = relu( BN2( conv9x1_t(y1) + conv_b ) + x )
// block = (n, 4 output t's): M = 4*32 = 128 padded rows = 8 M-tiles (1/wave);
// halo tile = (4+8)*32 = 384 rows staged via pure b128 copies (y1 is NHWC).
// 9-tap conv = 9 shifted-A GEMMs (row shift = dt*32) into the same C.
// ---------------------------------------------------------------------------
__global__ __launch_bounds__(256)
void k_tcn(const f16* __restrict__ y1t, const f16* __restrict__ cwh,
           const float* __restrict__ cb, const float* __restrict__ x,
           const float* __restrict__ g2, const float* __restrict__ b2,
           float* __restrict__ out) {
  __shared__ f16 yt[384][72];   // [halo'd padded pos][in-ch], 144B pitch

  int n   = blockIdx.y;
  int t0  = blockIdx.x * 4;     // 64 blocks cover T=256
  int tid = threadIdx.x;

  { // stage: 384 rows x 64 ch, one v8h per (row, 8-ch group)
    const f16* yb = y1t + (size_t)n * PP_ * COUT_;
    #pragma unroll
    for (int it = 0; it < 12; ++it) {
      int idx = it * 256 + tid;               // 3072 = 384*8
      int lr = idx >> 3, c8 = (idx & 7) * 8;
      int t = t0 - 4 + (lr >> 5);
      int v = lr & 31;
      v8h val = {};
      if (t >= 0 && t < T_)
        val = *(const v8h*)(yb + ((size_t)t * VP_ + v) * COUT_ + c8);
      *(v8h*)&yt[lr][c8] = val;
    }
  }
  __syncthreads();

  int wave = tid >> 5, lane = tid & 31, hi = lane >> 4, ln = lane & 15;
  int mt = wave;                 // one 16-row M-tile per wave

  v8f acc[4];
  #pragma unroll
  for (int nt = 0; nt < 4; ++nt) acc[nt] = (v8f){};

  for (int dt = 0; dt < 9; ++dt) {
    #pragma unroll
    for (int kk = 0; kk < 2; ++kk) {
      v16h a = frag_a(&yt[dt * 32 + mt * 16 + ln][0], kk, hi);
      #pragma unroll
      for (int nt = 0; nt < 4; ++nt) {
        v16h bf = frag_lin(cwh + ((size_t)(((dt * 2 + kk) * 4 + nt)) * 32 + lane) * 16);
        acc[nt] = wmma_f16(a, bf, acc[nt]);
      }
    }
  }

  #pragma unroll
  for (int nt = 0; nt < 4; ++nt) {
    int o = nt * 16 + ln;
    float s2 = g2[o] * RSBN;
    float sh = cb[o] * s2 + b2[o];
    const float* xb = x   + ((size_t)n * COUT_ + o) * P_;
    float*       ob = out + ((size_t)n * COUT_ + o) * P_;
    #pragma unroll
    for (int r = 0; r < 8; ++r) {
      int m  = mt * 16 + r + (hi ? 8 : 0);    // 0..127
      int lt = m >> 5, v = m & 31;
      if (v < V_) {
        int p = (t0 + lt) * V_ + v;
        float val = acc[nt][r] * s2 + sh + xb[p];
        ob[p] = val > 0.f ? val : 0.f;
      }
    }
  }
}

// ---------------------------------------------------------------------------
extern "C" void kernel_launch(void* const* d_in, const int* in_sizes, int n_in,
                              void* d_out, int out_size, void* d_ws, size_t ws_size,
                              hipStream_t stream) {
  (void)in_sizes; (void)n_in; (void)out_size; (void)ws_size;
  const float* x     = (const float*)d_in[0];
  const float* dA    = (const float*)d_in[1];
  const float* W     = (const float*)d_in[2];
  const float* bconv = (const float*)d_in[3];
  const float* g0    = (const float*)d_in[4];
  const float* b0    = (const float*)d_in[5];
  const float* g1    = (const float*)d_in[6];
  const float* b1    = (const float*)d_in[7];
  const float* cw    = (const float*)d_in[8];
  const float* cb    = (const float*)d_in[9];
  const float* g2    = (const float*)d_in[10];
  const float* b2    = (const float*)d_in[11];
  float* out = (float*)d_out;

  char* ws = (char*)d_ws;
  float* nA  = (float*)ws;                      //     60,000 B
  f16*   nAh = (f16*)(ws + 65536);              //     49,152 B
  f16*   wbh = (f16*)(ws + 131072);             //     24,576 B
  f16*   cwh = (f16*)(ws + 196608);             //     73,728 B
  f16*   ykh = (f16*)(ws + 327680);             // 100,663,296 B
  f16*   y1t = (f16*)(ws + 327680 + (size_t)N_ * D_ * PP_ * 2); // 33,554,432 B

  k_norm_a    <<<dim3(3),          256, 0, stream>>>(dA, nA);
  k_prep_mp_b <<<dim3(96),         256, 0, stream>>>(nA, nAh);
  k_prep_gemm_w<<<dim3(48),        256, 0, stream>>>(W, wbh);
  k_prep_tcn_w<<<dim3(144),        256, 0, stream>>>(cw, cwh);
  k_zero_pad  <<<dim3(6144),       256, 0, stream>>>(ykh);
  k_gcn_gemm  <<<dim3(100, 32),    256, 0, stream>>>(x, wbh, bconv, g0, b0, ykh);
  k_gcn_mp    <<<dim3(2, 64, 32),  256, 0, stream>>>(ykh, nAh, x, g1, b1, y1t);
  k_tcn       <<<dim3(64, 32),     256, 0, stream>>>(y1t, cwh, cb, x, g2, b2, out);
}
